// curlEncodedClassifier_75342316307130
// MI455X (gfx1250) — matrix operand order repaired and checked by reading, not compiled, not run on hardware
//
#include <hip/hip_runtime.h>
#include <hip/hip_bf16.h>

// ---------------- Problem constants (from reference) ----------------
#define BB   16
#define TT   1500
#define KK   16
#define DD   512
#define HID  2048
#define OUTC 512
#define NROWS (BB * TT)          // 24000

typedef __attribute__((ext_vector_type(16))) __bf16 bf16x16;
typedef __attribute__((ext_vector_type(8)))  float  v8f;
typedef __attribute__((ext_vector_type(4)))  unsigned int u32x4;
typedef __attribute__((ext_vector_type(8)))  int i32x8;
typedef __attribute__((ext_vector_type(4)))  int i32x4;

#if defined(__has_builtin)
#if __has_builtin(__builtin_amdgcn_tensor_load_to_lds)
#define HAVE_TDM 1
#else
#define HAVE_TDM 0
#endif
#else
#define HAVE_TDM 0
#endif

// LDS tile geometry: 128 rows x 64 bf16 (128 B data per row) + 16 B pad -> 144 B stride
#define ROWB       144
#define TILE_BYTES (128 * ROWB)      // 18432 B per tile

// ---------------- helpers ----------------
__device__ __forceinline__ __bf16 to_bf16(float f) {
    union { float f; unsigned u; } a; a.f = f;
    unsigned r = a.u + 0x7FFFu + ((a.u >> 16) & 1u);   // round-to-nearest-even
    union { unsigned short s; __bf16 b; } o; o.s = (unsigned short)(r >> 16);
    return o.b;
}

// A-fragment (16x32 bf16 MxK) from an LDS row: chunks at d0 and d0+32 bytes
__device__ __forceinline__ bf16x16 lds_afrag(const char* rowp, int d0) {
    union { uint4 q[2]; bf16x16 v; } u;
    u.q[0] = *reinterpret_cast<const uint4*>(rowp + d0);
    u.q[1] = *reinterpret_cast<const uint4*>(rowp + d0 + 32);
    return u.v;
}
// B-fragment (32x16 bf16 KxN) from an LDS row: chunks at d0 and d0+32 bytes
__device__ __forceinline__ bf16x16 lds_bfrag(const char* rowp, int d0) {
    union { uint4 q[2]; bf16x16 v; } u;
    u.q[0] = *reinterpret_cast<const uint4*>(rowp + d0);
    u.q[1] = *reinterpret_cast<const uint4*>(rowp + d0 + 32);
    return u.v;
}

#if HAVE_TDM
// Issue one TDM 2D tile load: 64 (dim0, contiguous K) x 128 (dim1, rows) bf16,
// LDS padding: 4 DWORDs after every 32 DWORDs (row stride 144 B in LDS).
// D# layout per CDNA5 ISA. 6-arg builtin (clang-23 lane):
// (uint32x4 g0, int32x8 g1, int32x4 g2, int32x4 g3, int32x8 unused, i32 cpol)
__device__ __forceinline__ void tdm_load_tile(const __bf16* g, unsigned lds_off,
                                              unsigned dim0, unsigned dim1,
                                              unsigned stride_elems) {
    unsigned long long ga = (unsigned long long)(uintptr_t)g;
    u32x4 g0;
    g0[0] = 1u;                                            // count=1, not restore, no gather
    g0[1] = lds_off;                                       // lds_addr (bytes)
    g0[2] = (unsigned)ga;                                  // global_addr[31:0]
    g0[3] = (unsigned)((ga >> 32) & 0x01FFFFFFu) | (2u << 30); // addr[56:32] | type=2
    i32x8 g1;
    g1[0] = (int)((1u << 16)      // data_size = 2 bytes
                | (1u << 20)      // pad_enable
                | (4u << 22)      // pad_interval: 32 DWORDs (128 B)
                | (3u << 25));    // pad_amount:   4 DWORDs (16 B)
    g1[1] = (int)((dim0 & 0xFFFFu) << 16);                 // tensor_dim0[15:0] @ bits63:48
    g1[2] = (int)((dim0 >> 16) | ((dim1 & 0xFFFFu) << 16));// tensor_dim0[31:16] | tensor_dim1[15:0]
    g1[3] = (int)((dim1 >> 16) | (64u << 16));             // tensor_dim1[31:16] | tile_dim0=64
    g1[4] = (int)128u;                                     // tile_dim1=128, tile_dim2=0
    g1[5] = (int)stride_elems;                             // tensor_dim0_stride[31:0]
    g1[6] = 0;                                             // stride[47:32]=0, dim1_stride=0
    g1[7] = 0;
    i32x4 gz = {0, 0, 0, 0};
    i32x8 z8 = {0, 0, 0, 0, 0, 0, 0, 0};
    __builtin_amdgcn_tensor_load_to_lds(g0, g1, gz, gz, z8, 0);
}
#else
// Fallback: cooperative global->LDS copy (256 threads, 16 B chunks)
__device__ __forceinline__ void coop_load_tile(const __bf16* g, char* ldst,
                                               int dim1bound, int stride_elems, int tid) {
#pragma unroll
    for (int s = 0; s < 4; ++s) {
        int idx = tid + s * 256;         // 1024 chunks of 16 B
        int r = idx >> 3, c = idx & 7;
        uint4 v = {0, 0, 0, 0};
        if (r < dim1bound)
            v = *reinterpret_cast<const uint4*>(g + (size_t)r * stride_elems + c * 8);
        *reinterpret_cast<uint4*>(ldst + r * ROWB + c * 16) = v;
    }
}
#endif

// ---------------- kernel 1: feats[b,t,d] = sum_k w[b,t,k]*lat[k,b,t,d] -> bf16 ----------------
__global__ __launch_bounds__(256)
void feats_kernel(const float* __restrict__ w, const float* __restrict__ lat,
                  __bf16* __restrict__ X0) {
    int idx = blockIdx.x * blockDim.x + threadIdx.x;      // over NROWS * (DD/4)
    int n  = idx >> 7;                                    // DD/4 = 128
    int dv = (idx & 127) << 2;
    if (n >= NROWS) return;

    const float* wn = w + n * KK;
    float ax = 0.f, ay = 0.f, az = 0.f, aw = 0.f;
#pragma unroll
    for (int k = 0; k < KK; ++k) {
        float wk = wn[k];
        const float4 lv = *reinterpret_cast<const float4*>(
            lat + ((size_t)k * NROWS + n) * DD + dv);
        ax = fmaf(wk, lv.x, ax);
        ay = fmaf(wk, lv.y, ay);
        az = fmaf(wk, lv.z, az);
        aw = fmaf(wk, lv.w, aw);
    }
    union { __bf16 b[4]; uint2 u; } pk;
    pk.b[0] = to_bf16(ax); pk.b[1] = to_bf16(ay);
    pk.b[2] = to_bf16(az); pk.b[3] = to_bf16(aw);
    *reinterpret_cast<uint2*>(X0 + (size_t)n * DD + dv) = pk.u;
}

// ---------------- kernel 2: f32 -> bf16 weight conversion ----------------
__global__ __launch_bounds__(256)
void cvt_bf16(const float* __restrict__ in, __bf16* __restrict__ out, int n) {
    int i = blockIdx.x * blockDim.x + threadIdx.x;
    if (i < n) out[i] = to_bf16(in[i]);
}

// ---------------- kernel 3: C = relu(A * B^T + bias), WMMA bf16 + TDM double-buffered LDS ----
// A: Mrows x Kdim bf16 row-major; Bw: Ndim x Kdim bf16 row-major.
// Block = 256 threads = 8 waves (2M x 4N), block tile 128x128, K-step 64.
// Wave tile 64x32 = 4 M-frags x 2 N-frags; 16 WMMAs per wave per K-step.
__global__ __launch_bounds__(256)
void gemm_bf16_wmma(const __bf16* __restrict__ A, const __bf16* __restrict__ Bw,
                    const float* __restrict__ bias,
                    __bf16* __restrict__ Cb, float* __restrict__ Cf,
                    int Mrows, int Kdim, int Ndim, int relu) {
    // [buf0 A][buf0 B][buf1 A][buf1 B]
    __shared__ __attribute__((aligned(16))) char lds_mem[4 * TILE_BYTES];

    const int lane  = threadIdx.x & 31;
    const int wid   = threadIdx.x >> 5;
    const int waveM = wid >> 2;
    const int waveN = wid & 3;
    const int half  = lane >> 4;
    const int lidx  = lane & 15;

    const int rowBlock = blockIdx.x * 128;
    const int colBlock = blockIdx.y * 128;

    // Per-wave LDS row offsets (within a tile)
    const int aRow0 = (waveM * 64 +      lidx) * ROWB;
    const int aRow1 = (waveM * 64 + 16 + lidx) * ROWB;
    const int aRow2 = (waveM * 64 + 32 + lidx) * ROWB;
    const int aRow3 = (waveM * 64 + 48 + lidx) * ROWB;
    const int bRow0 = (waveN * 32 +      lidx) * ROWB;
    const int bRow1 = (waveN * 32 + 16 + lidx) * ROWB;

    v8f acc[4][2] = {};

#if HAVE_TDM
    const unsigned ldsBase = (unsigned)(uintptr_t)lds_mem;  // low 32 bits = LDS byte offset
    const unsigned rowsLeft = (unsigned)(Mrows - rowBlock); // TDM OOB rows read as zero
    const unsigned colsLeft = (unsigned)(Ndim - colBlock);

    // Prologue: wave 0 kicks off tile 0 into buffer 0
    if (wid == 0) {
        tdm_load_tile(A  + (size_t)rowBlock * Kdim, ldsBase,
                      (unsigned)Kdim, rowsLeft, (unsigned)Kdim);
        tdm_load_tile(Bw + (size_t)colBlock * Kdim, ldsBase + TILE_BYTES,
                      (unsigned)Kdim, colsLeft, (unsigned)Kdim);
    }
    const int nIter = Kdim >> 6;
    for (int it = 0; it < nIter; ++it) {
        const int cur = it & 1;
        if (wid == 0) __builtin_amdgcn_s_wait_tensorcnt((short)0);
        __syncthreads();                         // tiles[cur] ready for everyone
        if (wid == 0 && (it + 1) < nIter) {      // prefetch tiles[cur^1] via TDM
            const int kk = (it + 1) << 6;
            const unsigned nb = (unsigned)(cur ^ 1) * 2u * TILE_BYTES;
            tdm_load_tile(A  + (size_t)rowBlock * Kdim + kk, ldsBase + nb,
                          (unsigned)(Kdim - kk), rowsLeft, (unsigned)Kdim);
            tdm_load_tile(Bw + (size_t)colBlock * Kdim + kk, ldsBase + nb + TILE_BYTES,
                          (unsigned)(Kdim - kk), colsLeft, (unsigned)Kdim);
        }
        const char* aT = lds_mem + cur * 2 * TILE_BYTES;
        const char* bT = aT + TILE_BYTES;
#pragma unroll
        for (int ksub = 0; ksub < 2; ++ksub) {
            const int dA = ksub * 64 + half * 16;
            const int dB = ksub * 64 + half * 16;
            // Batch all fragment loads into independent values so the scheduler
            // can stage s_wait_dscnt and keep the WMMA pipe busy.
            bf16x16 b0 = lds_bfrag(bT + bRow0, dB);
            bf16x16 b1 = lds_bfrag(bT + bRow1, dB);
            bf16x16 a0 = lds_afrag(aT + aRow0, dA);
            bf16x16 a1 = lds_afrag(aT + aRow1, dA);
            bf16x16 a2 = lds_afrag(aT + aRow2, dA);
            bf16x16 a3 = lds_afrag(aT + aRow3, dA);
            acc[0][0] = __builtin_amdgcn_wmma_f32_16x16x32_bf16(false, a0, false, b0, (short)0, acc[0][0], false, false);
            acc[0][1] = __builtin_amdgcn_wmma_f32_16x16x32_bf16(false, a0, false, b1, (short)0, acc[0][1], false, false);
            acc[1][0] = __builtin_amdgcn_wmma_f32_16x16x32_bf16(false, a1, false, b0, (short)0, acc[1][0], false, false);
            acc[1][1] = __builtin_amdgcn_wmma_f32_16x16x32_bf16(false, a1, false, b1, (short)0, acc[1][1], false, false);
            acc[2][0] = __builtin_amdgcn_wmma_f32_16x16x32_bf16(false, a2, false, b0, (short)0, acc[2][0], false, false);
            acc[2][1] = __builtin_amdgcn_wmma_f32_16x16x32_bf16(false, a2, false, b1, (short)0, acc[2][1], false, false);
            acc[3][0] = __builtin_amdgcn_wmma_f32_16x16x32_bf16(false, a3, false, b0, (short)0, acc[3][0], false, false);
            acc[3][1] = __builtin_amdgcn_wmma_f32_16x16x32_bf16(false, a3, false, b1, (short)0, acc[3][1], false, false);
        }
        __syncthreads();                         // done reading tiles[cur] before overwrite
    }
#else
    // Fallback: single-buffered cooperative staging with plain global loads
    char* aT = lds_mem;
    char* bT = lds_mem + TILE_BYTES;
    const int tid = threadIdx.x;
    const int nIter = Kdim >> 6;
    for (int it = 0; it < nIter; ++it) {
        const int kk = it << 6;
        __syncthreads();
        coop_load_tile(A  + (size_t)rowBlock * Kdim + kk, aT, Mrows - rowBlock, Kdim, tid);
        coop_load_tile(Bw + (size_t)colBlock * Kdim + kk, bT, Ndim - colBlock, Kdim, tid);
        __syncthreads();
#pragma unroll
        for (int ksub = 0; ksub < 2; ++ksub) {
            const int dA = ksub * 64 + half * 16;
            const int dB = ksub * 64 + half * 16;
            bf16x16 b0 = lds_bfrag(bT + bRow0, dB);
            bf16x16 b1 = lds_bfrag(bT + bRow1, dB);
            bf16x16 a0 = lds_afrag(aT + aRow0, dA);
            bf16x16 a1 = lds_afrag(aT + aRow1, dA);
            bf16x16 a2 = lds_afrag(aT + aRow2, dA);
            bf16x16 a3 = lds_afrag(aT + aRow3, dA);
            acc[0][0] = __builtin_amdgcn_wmma_f32_16x16x32_bf16(false, a0, false, b0, (short)0, acc[0][0], false, false);
            acc[0][1] = __builtin_amdgcn_wmma_f32_16x16x32_bf16(false, a0, false, b1, (short)0, acc[0][1], false, false);
            acc[1][0] = __builtin_amdgcn_wmma_f32_16x16x32_bf16(false, a1, false, b0, (short)0, acc[1][0], false, false);
            acc[1][1] = __builtin_amdgcn_wmma_f32_16x16x32_bf16(false, a1, false, b1, (short)0, acc[1][1], false, false);
            acc[2][0] = __builtin_amdgcn_wmma_f32_16x16x32_bf16(false, a2, false, b0, (short)0, acc[2][0], false, false);
            acc[2][1] = __builtin_amdgcn_wmma_f32_16x16x32_bf16(false, a2, false, b1, (short)0, acc[2][1], false, false);
            acc[3][0] = __builtin_amdgcn_wmma_f32_16x16x32_bf16(false, a3, false, b0, (short)0, acc[3][0], false, false);
            acc[3][1] = __builtin_amdgcn_wmma_f32_16x16x32_bf16(false, a3, false, b1, (short)0, acc[3][1], false, false);
        }
    }
#endif

    // Epilogue: D layout (32-bit C/D 16x16): lane holds col n=lane&15,
    // vgpr r holds row m = r + 8*(lane>=16).
    const int rowWave = rowBlock + waveM * 64;
    const int colWave = colBlock + waveN * 32;
    const int mhalf = half * 8;
#pragma unroll
    for (int j = 0; j < 2; ++j) {
        const int col = colWave + j * 16 + lidx;
        const float bv = bias[col];
#pragma unroll
        for (int i = 0; i < 4; ++i) {
#pragma unroll
            for (int r = 0; r < 8; ++r) {
                int row = rowWave + i * 16 + mhalf + r;
                if (row < Mrows) {
                    float v = acc[i][j][r] + bv;
                    if (relu) v = fmaxf(v, 0.0f);
                    if (Cb) Cb[(size_t)row * Ndim + col] = to_bf16(v);
                    else    Cf[(size_t)row * Ndim + col] = v;
                }
            }
        }
    }
}

// ---------------- host launcher ----------------
extern "C" void kernel_launch(void* const* d_in, const int* in_sizes, int n_in,
                              void* d_out, int out_size, void* d_ws, size_t ws_size,
                              hipStream_t stream) {
    const float* w   = (const float*)d_in[0];
    const float* lat = (const float*)d_in[1];
    const float* W1  = (const float*)d_in[2];  const float* b1 = (const float*)d_in[3];
    const float* W2  = (const float*)d_in[4];  const float* b2 = (const float*)d_in[5];
    const float* W3  = (const float*)d_in[6];  const float* b3 = (const float*)d_in[7];
    const float* W4  = (const float*)d_in[8];  const float* b4 = (const float*)d_in[9];
    float* out = (float*)d_out;

    char* ws = (char*)d_ws;
    size_t off = 0;
    auto carve = [&](size_t bytes) -> void* {
        void* p = ws + off;
        off = (off + bytes + 255) & ~(size_t)255;
        return p;
    };

    __bf16* X0  = (__bf16*)carve((size_t)NROWS * DD  * 2);
    __bf16* X1  = (__bf16*)carve((size_t)NROWS * HID * 2);
    __bf16* X2  = (__bf16*)carve((size_t)NROWS * HID * 2);
    __bf16* W1b = (__bf16*)carve((size_t)HID  * DD  * 2);
    __bf16* W2b = (__bf16*)carve((size_t)HID  * HID * 2);
    __bf16* W3b = (__bf16*)carve((size_t)HID  * HID * 2);
    __bf16* W4b = (__bf16*)carve((size_t)OUTC * HID * 2);

    // Weight conversions (f32 -> bf16)
    {
        int n1 = HID * DD, n2 = HID * HID, n4 = OUTC * HID;
        cvt_bf16<<<(n1 + 255) / 256, 256, 0, stream>>>(W1, W1b, n1);
        cvt_bf16<<<(n2 + 255) / 256, 256, 0, stream>>>(W2, W2b, n2);
        cvt_bf16<<<(n2 + 255) / 256, 256, 0, stream>>>(W3, W3b, n2);
        cvt_bf16<<<(n4 + 255) / 256, 256, 0, stream>>>(W4, W4b, n4);
    }

    // Mixture einsum -> bf16 activations X0
    {
        int total = NROWS * (DD / 4);
        feats_kernel<<<(total + 255) / 256, 256, 0, stream>>>(w, lat, X0);
    }

    const int mblocks = (NROWS + 127) / 128;   // 188

    // Layer 1: X0 (K=512) -> X1 (2048), relu
    gemm_bf16_wmma<<<dim3(mblocks, HID / 128), 256, 0, stream>>>(
        X0, W1b, b1, X1, nullptr, NROWS, DD, HID, 1);
    // Layer 2: X1 (K=2048) -> X2 (2048), relu
    gemm_bf16_wmma<<<dim3(mblocks, HID / 128), 256, 0, stream>>>(
        X1, W2b, b2, X2, nullptr, NROWS, HID, HID, 1);
    // Layer 3: X2 (K=2048) -> X1 (2048), relu
    gemm_bf16_wmma<<<dim3(mblocks, HID / 128), 256, 0, stream>>>(
        X2, W3b, b3, X1, nullptr, NROWS, HID, HID, 1);
    // Layer 4: X1 (K=2048) -> out (512) f32, no relu
    gemm_bf16_wmma<<<dim3(mblocks, OUTC / 128), 256, 0, stream>>>(
        X1, W4b, b4, nullptr, out, NROWS, HID, OUTC, 0);
}
